// MambaSubBlock_88338887344437
// MI455X (gfx1250) — matrix operand order repaired and checked by reading, not compiled, not run on hardware
//
#include <hip/hip_runtime.h>
#include <hip/hip_bf16.h>

// ---------------------------------------------------------------------------
// Mamba block for MI455X (gfx1250, wave32, WMMA, async-to-LDS staging).
// Shapes: B=8, L=2048, DIM=512, D_INNER=1024, D_STATE=16, DT_RANK=32, D_CONV=4
// ---------------------------------------------------------------------------

typedef __attribute__((ext_vector_type(16))) __bf16 bf16x16;
typedef __attribute__((ext_vector_type(8)))  float  f32x8;
typedef unsigned short ushort16_t;

#define M_ROWS  16384
#define LSEQ    2048
#define BSZ     8
#define DIMV    512
#define DINNER  1024
#define DSTATE  16
#define DTRANK  32
#define XZCOLS  2048
#define DBLCOLS 64

// f32 -> bf16 with round-to-nearest-even
__device__ __forceinline__ unsigned short f2bf(float f) {
    unsigned int u = __float_as_uint(f);
    unsigned int r = (u + 0x7fffu + ((u >> 16) & 1u)) >> 16;
    return (unsigned short)r;
}

__device__ __forceinline__ float silu_f(float x) {
    return x / (1.0f + __expf(-x));
}

__device__ __forceinline__ float softplus_f(float x) {
    return (x > 20.0f) ? x : log1pf(__expf(x));
}

// ---------------------------------------------------------------------------
// C[M,N] = A[M,K] * B[N,K]^T with A,B pre-converted bf16 row-major in global.
// Tiles staged via global_load_async_to_lds_b128 (ASYNCcnt DMA), double
// buffered; one v_wmma_f32_16x16x32_bf16 per 16x16 tile per 32-deep k-step.
// 256 threads = 8 waves arranged WGM x WGN.
// EPI: 0 = f32 store; 1 = softplus(acc + bias[n]) f32 store;
//      2 = f32 store + bf16 mirror store (Cbf).
// ---------------------------------------------------------------------------
template<int BM, int BN, int WGM, int WGN, int EPI>
__global__ __launch_bounds__(256)
void mamba_gemm_bf16(const unsigned short* __restrict__ Abf,
                     const unsigned short* __restrict__ Bbf,
                     float* __restrict__ Cg, const float* __restrict__ bias,
                     unsigned short* __restrict__ Cbf,
                     int K, int lda, int ldb, int ldc) {
    constexpr int BK  = 32;
    constexpr int WTM = BM / WGM;
    constexpr int WTN = BN / WGN;
    constexpr int FM  = WTM / 16;
    constexpr int FN  = WTN / 16;
    constexpr int LDT = 40;                 // 80B row: 64B data + 16B pad
    constexpr int CA  = (BM * 4) / 256;     // 16B chunks per thread (A tile)
    constexpr int CB  = (BN * 4) / 256;     // 16B chunks per thread (B tile)

    __shared__ alignas(16) unsigned short As[2][BM][LDT];
    __shared__ alignas(16) unsigned short Bs[2][BN][LDT];

    const int tid   = threadIdx.x;
    const int lane  = tid & 31;
    const int wid   = tid >> 5;
    const int wm    = wid / WGN;
    const int wn    = wid % WGN;
    const int wrow  = lane & 15;
    const int khalf = lane >> 4;

    const int m0 = blockIdx.y * BM;
    const int n0 = blockIdx.x * BN;

    // Issue async DMA of one (BM x 32 | BN x 32) bf16 tile pair into buffer.
    auto issue_tile = [&](int buf, int k0) {
#pragma unroll
        for (int i = 0; i < CA; ++i) {
            const int idx = tid + i * 256;
            const int r   = idx >> 2;
            const int c   = idx & 3;
            const unsigned lds =
                (unsigned)(unsigned long long)(const void*)&As[buf][r][c * 8];
            const unsigned long long g = (unsigned long long)
                (Abf + (size_t)(m0 + r) * lda + k0 + c * 8);
            asm volatile("global_load_async_to_lds_b128 %0, %1, off"
                         :: "v"(lds), "v"(g) : "memory");
        }
#pragma unroll
        for (int i = 0; i < CB; ++i) {
            const int idx = tid + i * 256;
            const int r   = idx >> 2;
            const int c   = idx & 3;
            const unsigned lds =
                (unsigned)(unsigned long long)(const void*)&Bs[buf][r][c * 8];
            const unsigned long long g = (unsigned long long)
                (Bbf + (size_t)(n0 + r) * ldb + k0 + c * 8);
            asm volatile("global_load_async_to_lds_b128 %0, %1, off"
                         :: "v"(lds), "v"(g) : "memory");
        }
    };

    f32x8 acc[FM][FN];
#pragma unroll
    for (int i = 0; i < FM; ++i)
#pragma unroll
        for (int j = 0; j < FN; ++j)
            acc[i][j] = (f32x8)0.0f;

    issue_tile(0, 0);                       // prologue

    for (int k0 = 0, it = 0; k0 < K; k0 += BK, ++it) {
        const int cur = it & 1;
        asm volatile("s_wait_asynccnt 0x0" ::: "memory");
        __syncthreads();
        if (k0 + BK < K) issue_tile(cur ^ 1, k0 + BK);

        // ---- fragments from LDS per wave32 WMMA layouts ----
        union AU { bf16x16 v; unsigned int u[8]; };
        AU af[FM], bf[FN];
#pragma unroll
        for (int fm = 0; fm < FM; ++fm) {
            const int ar = wm * WTM + fm * 16 + wrow;
#pragma unroll
            for (int j = 0; j < 8; ++j) {
                const int k = ((j & 4) ? 16 : 0) + khalf * 8 + 2 * (j & 3);
                af[fm].u[j] = *(const unsigned int*)&As[cur][ar][k];
            }
        }
#pragma unroll
        for (int fn = 0; fn < FN; ++fn) {
            const int bc = wn * WTN + fn * 16 + wrow;
#pragma unroll
            for (int j = 0; j < 8; ++j) {
                const int k = ((j & 4) ? 16 : 0) + khalf * 8 + 2 * (j & 3);
                bf[fn].u[j] = *(const unsigned int*)&Bs[cur][bc][k];
            }
        }

#pragma unroll
        for (int fm = 0; fm < FM; ++fm)
#pragma unroll
            for (int fn = 0; fn < FN; ++fn)
                acc[fm][fn] = __builtin_amdgcn_wmma_f32_16x16x32_bf16(
                    false, af[fm].v, false, bf[fn].v,
                    (short)0, acc[fm][fn], false, false);
    }

    // ---- epilogue: D layout VGPR r -> M = r + 8*khalf, N = wrow ----
#pragma unroll
    for (int fm = 0; fm < FM; ++fm) {
#pragma unroll
        for (int fn = 0; fn < FN; ++fn) {
            const int nn = n0 + wn * WTN + fn * 16 + wrow;
#pragma unroll
            for (int r = 0; r < 8; ++r) {
                const int mm = m0 + wm * WTM + fm * 16 + khalf * 8 + r;
                float v = acc[fm][fn][r];
                if (EPI == 1) v = softplus_f(v + bias[nn]);
                Cg[(size_t)mm * ldc + nn] = v;
                if (EPI == 2) Cbf[(size_t)mm * ldc + nn] = f2bf(v);
            }
        }
    }
}

// ---------------------------------------------------------------------------
// One-shot f32 -> bf16 conversion (inputs/weights).
// ---------------------------------------------------------------------------
__global__ __launch_bounds__(256)
void mamba_cvt_bf16(const float* __restrict__ in, unsigned short* __restrict__ out,
                    int n) {
    const int idx = blockIdx.x * 256 + threadIdx.x;
    if (idx < n) out[idx] = f2bf(in[idx]);
}

// ---------------------------------------------------------------------------
// Causal depthwise conv (k=4) + bias + SiLU; writes f32 (scan) + bf16 (GEMM).
// ---------------------------------------------------------------------------
__global__ __launch_bounds__(256)
void mamba_conv_silu(const float* __restrict__ xz, const float* __restrict__ cw,
                     const float* __restrict__ cb, float* __restrict__ u,
                     unsigned short* __restrict__ ubf) {
    const int idx = blockIdx.x * 256 + threadIdx.x;           // (ml, n)
    const int n   = idx & (DINNER - 1);
    const int ml  = idx >> 10;                                 // b*L + l
    const int l   = ml & (LSEQ - 1);
    float acc = cb[n];
#pragma unroll
    for (int k = 0; k < 4; ++k) {
        const int dl = k - 3;
        if (l + dl >= 0)
            acc += cw[n * 4 + k] * xz[(size_t)(ml + dl) * XZCOLS + n];
    }
    const float us = silu_f(acc);
    u[idx]   = us;
    ubf[idx] = f2bf(us);
}

// ---------------------------------------------------------------------------
// Selective scan: one thread per channel, 16 states in registers, loop over L.
// B_t/C_t staged through LDS (broadcast reads).
// ---------------------------------------------------------------------------
__global__ __launch_bounds__(256)
void mamba_scan(const float* __restrict__ dbl, const float* __restrict__ dt,
                const float* __restrict__ u, const float* __restrict__ A_log,
                float* __restrict__ ysc) {
    constexpr int TILE = 64;
    const int tid = threadIdx.x;
    const int b   = blockIdx.x >> 2;
    const int n   = ((blockIdx.x & 3) << 8) + tid;

    float An[DSTATE];
#pragma unroll
    for (int s = 0; s < DSTATE; ++s)
        An[s] = -__expf(A_log[n * DSTATE + s]);

    float h[DSTATE];
#pragma unroll
    for (int s = 0; s < DSTATE; ++s) h[s] = 0.0f;

    __shared__ float sBC[TILE][32];   // [l][0:16]=B_t, [16:32]=C_t

    for (int l0 = 0; l0 < LSEQ; l0 += TILE) {
        __syncthreads();
#pragma unroll
        for (int i = 0; i < (TILE * 32) / 256; ++i) {
            const int idx = tid + i * 256;
            const int lr  = idx >> 5;
            const int c   = idx & 31;
            sBC[lr][c] = dbl[(size_t)(b * LSEQ + l0 + lr) * DBLCOLS + DTRANK + c];
        }
        if (l0 + TILE < LSEQ)   // prefetch next B/C tile -> global_prefetch_b8
            __builtin_prefetch(dbl + (size_t)(b * LSEQ + l0 + TILE + (tid >> 2)) * DBLCOLS + DTRANK, 0, 3);
        __syncthreads();

        for (int lt = 0; lt < TILE; ++lt) {
            const size_t g = (size_t)(b * LSEQ + l0 + lt) * DINNER + n;
            const float dtv = dt[g];
            const float uv  = u[g];
            const float du  = dtv * uv;
            float y = 0.0f;
#pragma unroll
            for (int s = 0; s < DSTATE; ++s) {
                const float dA = __expf(dtv * An[s]);
                h[s] = dA * h[s] + du * sBC[lt][s];
                y   += h[s] * sBC[lt][DSTATE + s];
            }
            ysc[g] = y;
        }
    }
}

// ---------------------------------------------------------------------------
// ybf = bf16((yscan + u*D) * silu(z)); z = xz[:, 1024+n].
// ---------------------------------------------------------------------------
__global__ __launch_bounds__(256)
void mamba_gate(const float* __restrict__ xz, const float* __restrict__ u,
                const float* __restrict__ Dp, const float* __restrict__ ysc,
                unsigned short* __restrict__ ybf) {
    const int idx = blockIdx.x * 256 + threadIdx.x;
    const int n   = idx & (DINNER - 1);
    const int ml  = idx >> 10;
    const float z = xz[(size_t)ml * XZCOLS + DINNER + n];
    ybf[idx] = f2bf((ysc[idx] + u[idx] * Dp[n]) * silu_f(z));
}

// ---------------------------------------------------------------------------
extern "C" void kernel_launch(void* const* d_in, const int* in_sizes, int n_in,
                              void* d_out, int out_size, void* d_ws, size_t ws_size,
                              hipStream_t stream) {
    (void)in_sizes; (void)n_in; (void)out_size; (void)ws_size;

    const float* x       = (const float*)d_in[0];
    /* d_in[1] = mask (all ones for this problem) */
    const float* W_in    = (const float*)d_in[2];   // (2048, 512)
    const float* conv_w  = (const float*)d_in[3];   // (1024, 4)
    const float* conv_b  = (const float*)d_in[4];   // (1024,)
    const float* W_xproj = (const float*)d_in[5];   // (64, 1024)
    const float* W_dt    = (const float*)d_in[6];   // (1024, 32)
    const float* b_dt    = (const float*)d_in[7];   // (1024,)
    const float* A_log   = (const float*)d_in[8];   // (1024, 16)
    const float* Dp      = (const float*)d_in[9];   // (1024,)
    const float* W_out   = (const float*)d_in[10];  // (512, 1024)
    float*       out     = (float*)d_out;           // (16384, 512)

    // ---- workspace layout: f32 regions then bf16 regions ----
    float* xz  = (float*)d_ws;                          // 16384 x 2048
    float* u   = xz  + (size_t)M_ROWS * XZCOLS;         // 16384 x 1024
    float* dbl = u   + (size_t)M_ROWS * DINNER;         // 16384 x 64
    float* dt  = dbl + (size_t)M_ROWS * DBLCOLS;        // 16384 x 1024
    float* ysc = dt  + (size_t)M_ROWS * DINNER;         // 16384 x 1024

    unsigned short* xbf    = (unsigned short*)(ysc + (size_t)M_ROWS * DINNER);
    unsigned short* winbf  = xbf   + (size_t)M_ROWS * DIMV;          // 2048x512
    unsigned short* ubf    = winbf + (size_t)XZCOLS * DIMV;          // 16384x1024
    unsigned short* wxbf   = ubf   + (size_t)M_ROWS * DINNER;        // 64x1024
    unsigned short* dblbf  = wxbf  + (size_t)DBLCOLS * DINNER;       // 16384x64
    unsigned short* wdtbf  = dblbf + (size_t)M_ROWS * DBLCOLS;       // 1024x32
    unsigned short* ybf    = wdtbf + (size_t)DINNER * DTRANK;        // 16384x1024
    unsigned short* woutbf = ybf   + (size_t)M_ROWS * DINNER;        // 512x1024

    // 0) one-shot bf16 conversions of x and weights
    mamba_cvt_bf16<<<(M_ROWS * DIMV) / 256, 256, 0, stream>>>(x, xbf, M_ROWS * DIMV);
    mamba_cvt_bf16<<<(XZCOLS * DIMV) / 256, 256, 0, stream>>>(W_in, winbf, XZCOLS * DIMV);
    mamba_cvt_bf16<<<(DBLCOLS * DINNER) / 256, 256, 0, stream>>>(W_xproj, wxbf, DBLCOLS * DINNER);
    mamba_cvt_bf16<<<(DINNER * DTRANK) / 256, 256, 0, stream>>>(W_dt, wdtbf, DINNER * DTRANK);
    mamba_cvt_bf16<<<(DIMV * DINNER) / 256, 256, 0, stream>>>(W_out, woutbf, DIMV * DINNER);

    // 1) xz = x @ W_in^T     (M=16384, N=2048, K=512)
    mamba_gemm_bf16<128, 128, 2, 4, 0><<<dim3(XZCOLS / 128, M_ROWS / 128), 256, 0, stream>>>(
        xbf, winbf, xz, nullptr, nullptr, DIMV, DIMV, DIMV, XZCOLS);

    // 2) u = silu(causal_conv(xz[:, :1024]) + conv_b)  (f32 + bf16)
    mamba_conv_silu<<<(M_ROWS * DINNER) / 256, 256, 0, stream>>>(xz, conv_w, conv_b, u, ubf);

    // 3) dbl = u @ W_xproj^T (M=16384, N=64, K=1024), f32 + bf16 mirror
    mamba_gemm_bf16<128, 64, 4, 2, 2><<<dim3(1, M_ROWS / 128), 256, 0, stream>>>(
        ubf, wxbf, dbl, nullptr, dblbf, DINNER, DINNER, DINNER, DBLCOLS);

    // 4) dt = softplus(dbl[:, :32] @ W_dt^T + b_dt)  (N=1024, K=32)
    mamba_gemm_bf16<128, 128, 2, 4, 1><<<dim3(DINNER / 128, M_ROWS / 128), 256, 0, stream>>>(
        dblbf, wdtbf, dt, b_dt, nullptr, DTRANK, DBLCOLS, DTRANK, DINNER);

    // 5) selective scan -> ysc
    mamba_scan<<<BSZ * (DINNER / 256), 256, 0, stream>>>(dbl, dt, u, A_log, ysc);

    // 6) ybf = bf16((ysc + u*D) * silu(z))
    mamba_gate<<<(M_ROWS * DINNER) / 256, 256, 0, stream>>>(xz, u, Dp, ysc, ybf);

    // 7) out = y @ W_out^T (M=16384, N=512, K=1024)
    mamba_gemm_bf16<128, 128, 2, 4, 0><<<dim3(DIMV / 128, M_ROWS / 128), 256, 0, stream>>>(
        ybf, woutbf, out, nullptr, nullptr, DINNER, DINNER, DINNER, DIMV);
}
